// MultiHeadAttention_2774548873402
// MI455X (gfx1250) — matrix operand order repaired and checked by reading, not compiled, not run on hardware
//
#include <hip/hip_runtime.h>

// ---------------------------------------------------------------------------
// MHA forward for MI455X (gfx1250): bf16 WMMA + Tensor Data Mover staging.
// B=2, T=2048, D=1024, H=16, Dh=64.
// ---------------------------------------------------------------------------

typedef __bf16 bf16;
typedef __attribute__((ext_vector_type(16))) __bf16 v16bf;
typedef __attribute__((ext_vector_type(8)))  __bf16 v8bf;
typedef __attribute__((ext_vector_type(8)))  float  v8f;
typedef __attribute__((ext_vector_type(4)))  unsigned int u32x4;
typedef __attribute__((ext_vector_type(8)))  int i32x8;
typedef __attribute__((ext_vector_type(4)))  int i32x4;

#define BB 2
#define TT 2048
#define DD 1024
#define HH 16
#define DH 64

// ------------------------- elementwise converters --------------------------

__global__ void k_f32_to_bf16(const float* __restrict__ src,
                              bf16* __restrict__ dst, int n) {
    int i = blockIdx.x * blockDim.x + threadIdx.x;
    if (i < n) dst[i] = (bf16)src[i];
}

// Wt[n][k] = W[k][n]  (1024x1024), bf16 out.
__global__ void k_transpose_bf16(const float* __restrict__ W,
                                 bf16* __restrict__ Wt) {
    int i = blockIdx.x * blockDim.x + threadIdx.x;
    int n = i >> 10, k = i & 1023;
    Wt[i] = (bf16)W[k * DD + n];
}

// RoPE on Q,K and relayout [b][t][h*64+d] -> [b][h][t][d] (bf16)
__global__ void k_rope_relayout(const float* __restrict__ qf,
                                const float* __restrict__ kf,
                                const float* __restrict__ cosT,
                                const float* __restrict__ sinT,
                                bf16* __restrict__ qb,
                                bf16* __restrict__ kb) {
    int idx = blockIdx.x * blockDim.x + threadIdx.x;   // B*T*H*32 = 2^21
    int i = idx & 31;
    int h = (idx >> 5) & 15;
    int t = (idx >> 9) & 2047;
    int b = idx >> 20;
    float c = cosT[t * 32 + i];
    float s = sinT[t * 32 + i];
    int src = ((b * TT + t) * DD) + h * DH + 2 * i;
    int dst = (((b * HH + h) * TT + t) * DH) + 2 * i;
    {
        float tr = qf[src], ti = qf[src + 1];
        qb[dst]     = (bf16)(tr * c - ti * s);
        qb[dst + 1] = (bf16)(tr * s + ti * c);
    }
    {
        float tr = kf[src], ti = kf[src + 1];
        kb[dst]     = (bf16)(tr * c - ti * s);
        kb[dst + 1] = (bf16)(tr * s + ti * c);
    }
}

// V relayout [b][t][h*64+d] -> Vt[b][h][d][t] (bf16) for contiguous B-frags
__global__ void k_v_relayout(const float* __restrict__ vf,
                             bf16* __restrict__ vT) {
    int idx = blockIdx.x * blockDim.x + threadIdx.x;   // 2^22
    int d = idx & 63;
    int h = (idx >> 6) & 15;
    int t = (idx >> 10) & 2047;
    int b = idx >> 21;
    vT[(((size_t)(b * HH + h) * DH + d) * TT) + t] =
        (bf16)vf[((size_t)(b * TT + t) * DD) + h * DH + d];
}

// ----------------------- Tensor Data Mover helper --------------------------
// Issue TENSOR_LOAD_TO_LDS of a 2-D tile: tile_k (contiguous dim-0 elements,
// bf16) x tile_rows, row stride = stride0 elements.  D# per ISA 8.3/8.4.

__device__ __forceinline__ void tdm_load_2d(const bf16* gbase,
                                            unsigned ldsOff,
                                            unsigned tile_k,
                                            unsigned tile_rows,
                                            unsigned tensor_d0,
                                            unsigned tensor_d1,
                                            unsigned long long stride0) {
    unsigned long long ga = (unsigned long long)(size_t)gbase;
    u32x4 g0;
    g0[0] = 1u;                                        // count=1 (valid D#)
    g0[1] = ldsOff;                                    // lds_addr (bytes)
    g0[2] = (unsigned)(ga & 0xFFFFFFFFu);              // global_addr[31:0]
    g0[3] = (unsigned)((ga >> 32) & 0x01FFFFFFu)       // global_addr[56:32]
          | (2u << 30);                                // type=2 ("image")
    i32x8 g1;
    g1[0] = (int)(1u << 16);                           // data_size=2B
    g1[1] = (int)((tensor_d0 & 0xFFFFu) << 16);        // tensor_dim0[15:0]
    g1[2] = (int)((tensor_d0 >> 16) | ((tensor_d1 & 0xFFFFu) << 16));
    g1[3] = (int)((tensor_d1 >> 16) | (tile_k << 16)); // tile_dim0
    g1[4] = (int)tile_rows;                            // tile_dim1 (dim2=0)
    g1[5] = (int)(stride0 & 0xFFFFFFFFull);            // dim0_stride[31:0]
    g1[6] = (int)((stride0 >> 32) & 0xFFFFull);        // dim0_stride[47:32]
    g1[7] = 0;
    i32x4 z4;
    z4[0] = 0; z4[1] = 0; z4[2] = 0; z4[3] = 0;
    i32x8 z8;
#pragma unroll
    for (int e = 0; e < 8; ++e) z8[e] = 0;
    __builtin_amdgcn_tensor_load_to_lds(g0, g1, z4, z4, z8, 0);
}

// ------------------------------ WMMA GEMM ----------------------------------
// C[M][N] (fp32) = A[M][K] (bf16) * Bt[N][K]^T (bf16).
// Block = 4 waves -> 64x64 tile; A/B chunks TDM-staged in LDS, double buffer.

__global__ void k_gemm_tdm(const bf16* __restrict__ A,
                           const bf16* __restrict__ Bt,
                           float* __restrict__ C,
                           int M, int N, int K) {
    __shared__ __align__(32) bf16 sA[2][64][32];   // [buf][m][k]  4 KB each
    __shared__ __align__(32) bf16 sB[2][64][32];   // [buf][n][k]  4 KB each

    const int lane = threadIdx.x & 31;
    const int wid  = threadIdx.x >> 5;
    const int ngB  = N >> 6;
    const int mB   = (blockIdx.x / ngB) * 64;      // block row base
    const int nB   = (blockIdx.x % ngB) * 64;      // block col base

    const int r   = lane & 15;
    const int hi  = lane >> 4;
    const int hi8 = hi * 8;

    const int nch = K >> 5;                        // K/32 chunks

    if (wid == 0) {
        tdm_load_2d(A + (size_t)mB * K, (unsigned)(size_t)&sA[0][0][0],
                    32, 64, (unsigned)K, (unsigned)M, (unsigned)K);
        tdm_load_2d(Bt + (size_t)nB * K, (unsigned)(size_t)&sB[0][0][0],
                    32, 64, (unsigned)K, (unsigned)N, (unsigned)K);
    }

    v8f acc[4];
#pragma unroll
    for (int t = 0; t < 4; ++t)
#pragma unroll
        for (int e = 0; e < 8; ++e) acc[t][e] = 0.0f;

    for (int i = 0; i < nch; ++i) {
        const int buf = i & 1;
        if (wid == 0) {
            if (i + 1 < nch) {
                const int k0 = (i + 1) * 32;
                tdm_load_2d(A + (size_t)mB * K + k0,
                            (unsigned)(size_t)&sA[buf ^ 1][0][0],
                            32, 64, (unsigned)K, (unsigned)M, (unsigned)K);
                tdm_load_2d(Bt + (size_t)nB * K + k0,
                            (unsigned)(size_t)&sB[buf ^ 1][0][0],
                            32, 64, (unsigned)K, (unsigned)N, (unsigned)K);
                __builtin_amdgcn_s_wait_tensorcnt(2);  // chunk i complete
            } else {
                __builtin_amdgcn_s_wait_tensorcnt(0);  // last chunk complete
            }
        }
        __syncthreads();

        // A fragment: contiguous k = 16*hi + s (matches B)
        v16bf a;
        {
            const bf16* row = &sA[buf][wid * 16 + r][0];
            a = *(const v16bf*)(row + hi * 16);
        }
#pragma unroll
        for (int t = 0; t < 4; ++t) {
            const v16bf b = *(const v16bf*)&sB[buf][t * 16 + r][hi * 16];
            acc[t] = __builtin_amdgcn_wmma_f32_16x16x32_bf16(
                false, a, false, b, (short)0, acc[t], false, false);
        }
        __syncthreads();
    }

#pragma unroll
    for (int t = 0; t < 4; ++t)
#pragma unroll
        for (int j = 0; j < 8; ++j)
            C[(size_t)(mB + wid * 16 + j + hi8) * N + nB + t * 16 + r] =
                acc[t][j];
}

// --------------------------- flash attention -------------------------------
// 1 wave = one (b,h) x 16-row Q tile.  Online softmax, causal.

__device__ __forceinline__ float rowMax16(float v) {
    v = fmaxf(v, __shfl_xor(v, 1, 32));
    v = fmaxf(v, __shfl_xor(v, 2, 32));
    v = fmaxf(v, __shfl_xor(v, 4, 32));
    v = fmaxf(v, __shfl_xor(v, 8, 32));
    return v;
}
__device__ __forceinline__ float rowSum16(float v) {
    v += __shfl_xor(v, 1, 32);
    v += __shfl_xor(v, 2, 32);
    v += __shfl_xor(v, 4, 32);
    v += __shfl_xor(v, 8, 32);
    return v;
}

__global__ void k_attention(const bf16* __restrict__ qb,
                            const bf16* __restrict__ kb,
                            const bf16* __restrict__ vT,
                            bf16* __restrict__ ao) {
    __shared__ __align__(32) bf16 pTile[4][16][32];

    const int lane = threadIdx.x & 31;
    const int wid  = threadIdx.x >> 5;
    const int w    = blockIdx.x * (blockDim.x >> 5) + wid;
    if (w >= BB * HH * (TT / 16)) return;

    const int bh = w >> 7;           // T/16 = 128 tiles
    const int mt = w & 127;
    const int b  = bh >> 4;
    const int h  = bh & 15;

    const int r   = lane & 15;
    const int hi  = lane >> 4;
    const int hi8 = hi * 8;

    const bf16* Q  = qb + (size_t)bh * TT * DH;
    const bf16* Kp = kb + (size_t)bh * TT * DH;
    const bf16* V  = vT + (size_t)bh * DH * TT;

    v16bf aq[2];
    {
        const bf16* qRow = Q + (size_t)(mt * 16 + r) * DH + hi * 16;
#pragma unroll
        for (int d = 0; d < 2; ++d) {
            aq[d] = *(const v16bf*)(qRow + d * 32);
        }
    }

    v8f o[4];
#pragma unroll
    for (int t = 0; t < 4; ++t)
#pragma unroll
        for (int e = 0; e < 8; ++e) o[t][e] = 0.0f;

    float m_i[8], l_i[8];
#pragma unroll
    for (int j = 0; j < 8; ++j) { m_i[j] = -1e30f; l_i[j] = 0.0f; }

    const float scale = 0.125f;               // 1/sqrt(64)
    const int rowmax  = mt * 16 + 15;
    const int nchunks = rowmax / 32 + 1;

    for (int c = 0; c < nchunks; ++c) {
        const int c0 = c * 32;

        v8f s[2];
#pragma unroll
        for (int nt = 0; nt < 2; ++nt) {
#pragma unroll
            for (int e = 0; e < 8; ++e) s[nt][e] = 0.0f;
#pragma unroll
            for (int d = 0; d < 2; ++d) {
                const v16bf bk =
                    *(const v16bf*)(Kp + (size_t)(c0 + nt * 16 + r) * DH +
                                    d * 32 + hi * 16);
                s[nt] = __builtin_amdgcn_wmma_f32_16x16x32_bf16(
                    false, aq[d], false, bk, (short)0, s[nt], false, false);
            }
        }

#pragma unroll
        for (int j = 0; j < 8; ++j) {
            const int row  = mt * 16 + j + hi8;
            const int col0 = c0 + r;
            const int col1 = c0 + 16 + r;
            float s0 = (col0 <= row) ? s[0][j] * scale : -1e30f;
            float s1 = (col1 <= row) ? s[1][j] * scale : -1e30f;

            float cm = rowMax16(fmaxf(s0, s1));
            float nm = fmaxf(m_i[j], cm);
            float alpha = __expf(m_i[j] - nm);
            m_i[j] = nm;

            s0 = __expf(s0 - nm);
            s1 = __expf(s1 - nm);
            float rs = rowSum16(s0 + s1);
            l_i[j] = l_i[j] * alpha + rs;
#pragma unroll
            for (int t = 0; t < 4; ++t) o[t][j] *= alpha;

            pTile[wid][j + hi8][r]      = (bf16)s0;
            pTile[wid][j + hi8][16 + r] = (bf16)s1;
        }

        // wave-local D-layout -> A-layout relayout through LDS
        asm volatile("s_wait_dscnt 0" ::: "memory");

        v16bf ap;
        {
            ap = *(const v16bf*)&pTile[wid][r][hi * 16];
        }

#pragma unroll
        for (int t = 0; t < 4; ++t) {
            const v16bf bv =
                *(const v16bf*)(V + (size_t)(t * 16 + r) * TT + c0 + hi * 16);
            o[t] = __builtin_amdgcn_wmma_f32_16x16x32_bf16(
                false, ap, false, bv, (short)0, o[t], false, false);
        }
    }

#pragma unroll
    for (int j = 0; j < 8; ++j) {
        const float inv = 1.0f / l_i[j];
        const int row = mt * 16 + j + hi8;
#pragma unroll
        for (int t = 0; t < 4; ++t)
            ao[(size_t)(b * TT + row) * DD + h * DH + t * 16 + r] =
                (bf16)(o[t][j] * inv);
    }
}

// ------------------------------ host side ----------------------------------

extern "C" void kernel_launch(void* const* d_in, const int* in_sizes, int n_in,
                              void* d_out, int out_size, void* d_ws,
                              size_t ws_size, hipStream_t stream) {
    (void)in_sizes; (void)n_in; (void)out_size; (void)ws_size;

    const float* x    = (const float*)d_in[0];
    const float* wq   = (const float*)d_in[1];
    const float* wk   = (const float*)d_in[2];
    const float* wv   = (const float*)d_in[3];
    const float* wo   = (const float*)d_in[4];
    const float* fcos = (const float*)d_in[5];
    const float* fsin = (const float*)d_in[6];
    float* out = (float*)d_out;

    char* ws = (char*)d_ws;
    const size_t MB = 1u << 20;
    bf16* xb  = (bf16*)(ws + 0 * MB);    // 8 MB   (4096x1024 bf16)
    bf16* wqT = (bf16*)(ws + 8 * MB);    // 2 MB
    bf16* wkT = (bf16*)(ws + 10 * MB);   // 2 MB
    bf16* wvT = (bf16*)(ws + 12 * MB);   // 2 MB
    bf16* woT = (bf16*)(ws + 14 * MB);   // 2 MB
    float* qf = (float*)(ws + 16 * MB);  // 16 MB  (4096x1024 f32)
    float* kf = (float*)(ws + 32 * MB);  // 16 MB
    float* vf = (float*)(ws + 48 * MB);  // 16 MB
    bf16* qb  = (bf16*)(ws + 64 * MB);   // 8 MB   [b][h][t][64]
    bf16* kb  = (bf16*)(ws + 72 * MB);   // 8 MB
    bf16* vT  = (bf16*)(ws + 80 * MB);   // 8 MB   [b][h][64][t]
    bf16* ao  = (bf16*)(ws + 88 * MB);   // 8 MB   [b][t][1024]

    const int M = BB * TT;               // 4096

    k_f32_to_bf16<<<(M * DD + 255) / 256, 256, 0, stream>>>(x, xb, M * DD);

    k_transpose_bf16<<<(DD * DD) / 256, 256, 0, stream>>>(wq, wqT);
    k_transpose_bf16<<<(DD * DD) / 256, 256, 0, stream>>>(wk, wkT);
    k_transpose_bf16<<<(DD * DD) / 256, 256, 0, stream>>>(wv, wvT);
    k_transpose_bf16<<<(DD * DD) / 256, 256, 0, stream>>>(wo, woT);

    // QKV projections: block = 64x64 tile, TDM-staged LDS
    const int gemmBlocks = (M / 64) * (DD / 64);     // 1024
    k_gemm_tdm<<<gemmBlocks, 128, 0, stream>>>(xb, wqT, qf, M, DD, DD);
    k_gemm_tdm<<<gemmBlocks, 128, 0, stream>>>(xb, wkT, kf, M, DD, DD);
    k_gemm_tdm<<<gemmBlocks, 128, 0, stream>>>(xb, wvT, vf, M, DD, DD);

    k_rope_relayout<<<(BB * TT * HH * 32) / 256, 256, 0, stream>>>(
        qf, kf, fcos, fsin, qb, kb);
    k_v_relayout<<<(BB * TT * DD) / 256, 256, 0, stream>>>(vf, vT);

    k_attention<<<BB * HH * (TT / 16) / 4, 128, 0, stream>>>(qb, kb, vT, ao);

    k_gemm_tdm<<<gemmBlocks, 128, 0, stream>>>(ao, woT, out, M, DD, DD);
}
